// MHPLSTMCore_17454747090984
// MI455X (gfx1250) — matrix operand order repaired and checked by reading, not compiled, not run on hardware
//
#include <hip/hip_runtime.h>

// ---------------------------------------------------------------------------
// MHP-LSTM core for MI455X (gfx1250, wave32, WMMA).
//
// Shapes: B=4, S=4096, H=8, D=128. Rows NR = B*S = 16384.
// Memory-bound (~1 GB traffic @ 23.3 TB/s ~ 45us; only ~34 GF of GEMM).
// GEMMs use v_wmma_f32_16x16x32_f16. Weights are pre-swizzled into per-lane
// WMMA-fragment order so every B fragment is ONE contiguous 32B v16h load;
// A fragments are two ds_read_b128 from an LDS-staged tile. The two
// sequential S-recurrences are chunked (32 chunks of 128) via linear-
// recurrence composition for 131072-thread parallelism.
// ---------------------------------------------------------------------------

#define BATCH 4
#define SEQ   4096
#define NH    8
#define DH    128
#define NR    (BATCH * SEQ)   // 16384
#define HD    (NH * DH)       // 1024
#define K2D   (2 * DH)        // 256 (GEMM K)
#define N3D   (3 * DH)        // 384 (hid GEMM N per head)
#define NC    32              // scan chunks
#define CL    (SEQ / NC)      // 128
#define EPS   1e-6f

typedef __attribute__((ext_vector_type(16))) _Float16 v16h;
typedef __attribute__((ext_vector_type(8)))  _Float16 v8h;
typedef __attribute__((ext_vector_type(8)))  float    v8f;

__device__ __forceinline__ float sigmoidf_(float x) {
    return 1.0f / (1.0f + __expf(-x));
}

// ------------------------------------------------ weight swizzle (f32->f16)
// Output layout: fragment-contiguous per lane:
//   Wsw[(((h*NT + nt)*8 + kk)*32 + lane)*16 + e] = W[h][k][n]
//   with k = kk*32 + (lane>>4)*8 + (e&7) + ((e>>3)<<4), n = nt*16 + (lane&15).
// Each lane's 16 halves for one WMMA B-fragment are then one 32B run.
__global__ void k_swz_w(const float* __restrict__ W, _Float16* __restrict__ Wsw,
                        int NT, int Ncols) {
    int i = blockIdx.x * 256 + threadIdx.x;
    int total = NH * NT * 8 * 512;
    if (i >= total) return;
    int e    = i & 15;
    int lane = (i >> 4) & 31;
    int kk   = (i >> 9) & 7;
    int nt   = (i >> 12) % NT;
    int h    = (i >> 12) / NT;
    int k = kk * 32 + ((lane >> 4) << 3) + (e & 7) + ((e >> 3) << 4);
    int n = nt * 16 + (lane & 15);
    Wsw[i] = (_Float16)W[((size_t)h * K2D + k) * Ncols + n];
}

// ------------------------------------------------- chunked shifted cumsum
__global__ __launch_bounds__(256) void k_csum_chunk(
    const float* __restrict__ x, float* __restrict__ chunkS) {
    int tid = blockIdx.x * 256 + threadIdx.x;          // < B*NC*H*D = 131072
    int hd = tid & (HD - 1);
    int c  = (tid >> 10) & (NC - 1);
    int b  = tid >> 15;
    float acc = 0.0f;
    for (int sl = 0; sl < CL; ++sl) {
        int s = c * CL + sl;
        if (s > 0) acc += x[((size_t)b * SEQ + (s - 1)) * HD + hd];
    }
    chunkS[tid] = acc;
}

__global__ __launch_bounds__(256) void k_csum_scan(
    const float* __restrict__ chunkS, float* __restrict__ chunkOff) {
    int tid = blockIdx.x * 256 + threadIdx.x;          // < B*H*D = 4096
    int rem = tid & (HD - 1);
    int b   = tid >> 10;
    float run = 0.0f;
    for (int c = 0; c < NC; ++c) {
        size_t i = ((size_t)(b * NC + c) << 10) + rem;
        float v = chunkS[i];
        chunkOff[i] = run;
        run += v;
    }
}

__global__ __launch_bounds__(256) void k_csum_write(
    const float* __restrict__ x, const float* __restrict__ chunkOff,
    float* __restrict__ csum) {
    int tid = blockIdx.x * 256 + threadIdx.x;
    int hd = tid & (HD - 1);
    int c  = (tid >> 10) & (NC - 1);
    int b  = tid >> 15;
    float acc = chunkOff[tid];
    for (int sl = 0; sl < CL; ++sl) {
        int s = c * CL + sl;
        if (s > 0) acc += x[((size_t)b * SEQ + (s - 1)) * HD + hd];
        csum[((size_t)b * SEQ + s) * HD + hd] = acc;   // inclusive cumsum of shifted
    }
}

// ---------------------------------- LN(csum) over (h,d) + pack cat = [x|csum]
__global__ __launch_bounds__(256) void k_ln_csum_pack(
    const float* __restrict__ csum, const float* __restrict__ x,
    const float* __restrict__ g, const float* __restrict__ bb,
    _Float16* __restrict__ cat) {
    __shared__ float rs[256], rq[256];
    int r = blockIdx.x, t = threadIdx.x;
    size_t base = (size_t)r * HD;
    float v[4], s = 0.f, q = 0.f;
#pragma unroll
    for (int j = 0; j < 4; ++j) {
        v[j] = csum[base + t + 256 * j];
        s += v[j]; q += v[j] * v[j];
    }
    rs[t] = s; rq[t] = q; __syncthreads();
    for (int st = 128; st; st >>= 1) {
        if (t < st) { rs[t] += rs[t + st]; rq[t] += rq[t + st]; }
        __syncthreads();
    }
    float mean = rs[0] * (1.0f / HD);
    float var  = rq[0] * (1.0f / HD) - mean * mean;
    float rinv = rsqrtf(var + EPS);
#pragma unroll
    for (int j = 0; j < 4; ++j) {
        int flat = t + 256 * j;
        int h = flat >> 7, d = flat & 127;
        float ln = (v[j] - mean) * rinv * g[flat] + bb[flat];
        size_t o = (((size_t)r * NH + h) << 8) + d;    // (r,h,256) layout
        cat[o]      = (_Float16)x[base + flat];        // first 128: x
        cat[o + DH] = (_Float16)ln;                    // second 128: LN(csum)
    }
}

// ---------------------------------------------------------- WMMA fragments
// A (16-bit, 16x32): lane l: m=l&15, hl=l>>4; elem e -> k=(e&7)+((e>>3)<<4)+hl*8
//   => two contiguous 8-half runs at k0+hl*8 and k0+16+hl*8  (2x ds_read_b128)
// B: pre-swizzled, one contiguous v16h per lane.
// C/D f32: elem r -> M = r + (l>>4)*8, N = l&15.
__device__ __forceinline__ v16h load_afrag(const _Float16* rowp, int k0, int hl) {
    v8h lo = *(const v8h*)(rowp + k0 + hl * 8);
    v8h hi = *(const v8h*)(rowp + k0 + 16 + hl * 8);
    return __builtin_shufflevector(lo, hi, 0, 1, 2, 3, 4, 5, 6, 7,
                                   8, 9, 10, 11, 12, 13, 14, 15);
}

// -------------------------------------------------------- WMMA hid GEMM
__global__ __launch_bounds__(256) void k_hid_gemm(
    const _Float16* __restrict__ cat, const _Float16* __restrict__ Wsw,
    const float* __restrict__ bias, float* __restrict__ out) {
    __shared__ _Float16 At[16][264];                   // 16 x 256 (+pad, 16B-aligned rows)
    const int h = blockIdx.y;
    const size_t r0 = (size_t)blockIdx.x * 16;
    const int t = threadIdx.x;
    {   // cooperative A-tile stage (coalesced 32B per thread)
        int m = t >> 4, kb = (t & 15) << 4;
        const uint4* gp = (const uint4*)(cat + (((r0 + m) * NH + h) << 8) + kb);
        uint4 u0 = gp[0], u1 = gp[1];
        *(uint4*)(&At[m][kb])     = u0;
        *(uint4*)(&At[m][kb + 8]) = u1;
    }
    __syncthreads();
    const int lane = t & 31, w = t >> 5;
    const int hl = lane >> 4, mn = lane & 15;
    const _Float16* arow = &At[mn][0];
    const int nt0 = w, nt1 = w + 8, nt2 = w + 16;      // 24 n-tiles / 8 waves
    // fragment-order weight base for (h, nt, kk=0), this lane:
    const _Float16* bp0 = Wsw + ((size_t)(h * 24 + nt0) * 8 * 32 + lane) * 16;
    const _Float16* bp1 = Wsw + ((size_t)(h * 24 + nt1) * 8 * 32 + lane) * 16;
    const _Float16* bp2 = Wsw + ((size_t)(h * 24 + nt2) * 8 * 32 + lane) * 16;
    v8f acc0 = {}, acc1 = {}, acc2 = {};
#pragma unroll
    for (int kk = 0; kk < 8; ++kk) {
        v16h a = load_afrag(arow, kk * 32, hl);
        v16h b0 = *(const v16h*)(bp0 + (size_t)kk * 512);
        v16h b1 = *(const v16h*)(bp1 + (size_t)kk * 512);
        v16h b2 = *(const v16h*)(bp2 + (size_t)kk * 512);
        acc0 = __builtin_amdgcn_wmma_f32_16x16x32_f16(false, a, false, b0, (short)0, acc0, false, false);
        acc1 = __builtin_amdgcn_wmma_f32_16x16x32_f16(false, a, false, b1, (short)0, acc1, false, false);
        acc2 = __builtin_amdgcn_wmma_f32_16x16x32_f16(false, a, false, b2, (short)0, acc2, false, false);
    }
    const int n0 = nt0 * 16, n1 = nt1 * 16, n2 = nt2 * 16;
    const float bv0 = bias[h * N3D + n0 + mn];
    const float bv1 = bias[h * N3D + n1 + mn];
    const float bv2 = bias[h * N3D + n2 + mn];
#pragma unroll
    for (int ri = 0; ri < 8; ++ri) {
        int m = ri + hl * 8;
        size_t o = ((r0 + m) * NH + h) * (size_t)N3D;
        out[o + n0 + mn] = acc0[ri] + bv0;
        out[o + n1 + mn] = acc1[ri] + bv1;
        out[o + n2 + mn] = acc2[ri] + bv2;
    }
}

// --------------------------- LN(hid) over (h,3,d)=3072 + gate nonlinearities
__global__ __launch_bounds__(256) void k_ln_hid_gates(
    const float* __restrict__ hid, const float* __restrict__ g,
    const float* __restrict__ bb, float* __restrict__ fgate,
    float* __restrict__ igh) {
    __shared__ float rs[256], rq[256];
    __shared__ float sh[NH * N3D];                     // 3072 f32 = 12 KB
    int r = blockIdx.x, t = threadIdx.x;
    size_t base = (size_t)r * (NH * N3D);
    float v[12], s = 0.f, q = 0.f;
#pragma unroll
    for (int j = 0; j < 12; ++j) {
        v[j] = hid[base + t + 256 * j];
        s += v[j]; q += v[j] * v[j];
    }
    rs[t] = s; rq[t] = q; __syncthreads();
    for (int st = 128; st; st >>= 1) {
        if (t < st) { rs[t] += rs[t + st]; rq[t] += rq[t + st]; }
        __syncthreads();
    }
    float mean = rs[0] * (1.0f / (NH * N3D));
    float var  = rq[0] * (1.0f / (NH * N3D)) - mean * mean;
    float rinv = rsqrtf(var + EPS);
#pragma unroll
    for (int j = 0; j < 12; ++j) {
        int flat = t + 256 * j;
        sh[flat] = (v[j] - mean) * rinv * g[flat] + bb[flat];
    }
    __syncthreads();
    size_t ob = (size_t)r * HD;
#pragma unroll
    for (int j = 0; j < 4; ++j) {
        int p = t + 256 * j;
        int h = p >> 7, d = p & 127;
        float ig = sh[h * N3D + d];
        float fg = sh[h * N3D + DH + d];
        float hv = sh[h * N3D + 2 * DH + d];
        fgate[ob + p] = sigmoidf_(fg);
        igh[ob + p]   = sigmoidf_(ig) * fmaxf(hv, 0.0f);
    }
}

// ------------------------------------- chunked linear scan c = f*c + i
__global__ __launch_bounds__(256) void k_cell_chunk(
    const float* __restrict__ fgate, const float* __restrict__ igh,
    float* __restrict__ chunkA, float* __restrict__ chunkB) {
    int tid = blockIdx.x * 256 + threadIdx.x;
    int hd = tid & (HD - 1);
    int c  = (tid >> 10) & (NC - 1);
    int b  = tid >> 15;
    float A = 1.0f, Bv = 0.0f;
    for (int sl = 0; sl < CL; ++sl) {
        size_t i = ((size_t)b * SEQ + c * CL + sl) * HD + hd;
        float f = fgate[i], iv = igh[i];
        A = f * A;
        Bv = f * Bv + iv;
    }
    chunkA[tid] = A;
    chunkB[tid] = Bv;
}

__global__ __launch_bounds__(256) void k_cell_scan(
    const float* __restrict__ chunkA, const float* __restrict__ chunkB,
    const float* __restrict__ init_cx, float* __restrict__ chunkC) {
    int tid = blockIdx.x * 256 + threadIdx.x;          // < 4096
    int rem = tid & (HD - 1);
    int b   = tid >> 10;
    float cc = init_cx[rem];                           // (1,H,D) broadcast
    for (int c = 0; c < NC; ++c) {
        size_t i = ((size_t)(b * NC + c) << 10) + rem;
        chunkC[i] = cc;                                // chunk carry-in
        cc = chunkA[i] * cc + chunkB[i];
    }
}

__global__ __launch_bounds__(256) void k_cell_write(
    const float* __restrict__ fgate, const float* __restrict__ igh,
    const float* __restrict__ chunkC, float* __restrict__ cell,
    _Float16* __restrict__ cat) {
    int tid = blockIdx.x * 256 + threadIdx.x;
    int hd = tid & (HD - 1);
    int c  = (tid >> 10) & (NC - 1);
    int b  = tid >> 15;
    int h = hd >> 7, d = hd & 127;
    float cc = chunkC[tid];
    for (int sl = 0; sl < CL; ++sl) {
        size_t r = (size_t)b * SEQ + c * CL + sl;
        size_t i = r * HD + hd;
        cc = fgate[i] * cc + igh[i];
        cell[i] = cc;
        cat[((r * NH + h) << 8) + DH + d] = (_Float16)cc;  // overwrite csum half
    }
}

// -------------------------------------------------------- WMMA og GEMM
__global__ __launch_bounds__(256) void k_og_gemm(
    const _Float16* __restrict__ cat, const _Float16* __restrict__ Wsw,
    const float* __restrict__ bias, float* __restrict__ out) {
    __shared__ _Float16 At[16][264];
    const int h = blockIdx.y;
    const size_t r0 = (size_t)blockIdx.x * 16;
    const int t = threadIdx.x;
    {
        int m = t >> 4, kb = (t & 15) << 4;
        const uint4* gp = (const uint4*)(cat + (((r0 + m) * NH + h) << 8) + kb);
        uint4 u0 = gp[0], u1 = gp[1];
        *(uint4*)(&At[m][kb])     = u0;
        *(uint4*)(&At[m][kb + 8]) = u1;
    }
    __syncthreads();
    const int lane = t & 31, w = t >> 5;
    const int hl = lane >> 4, mn = lane & 15;
    const _Float16* arow = &At[mn][0];
    const _Float16* bp = Wsw + ((size_t)(h * 8 + w) * 8 * 32 + lane) * 16;
    v8f acc = {};
#pragma unroll
    for (int kk = 0; kk < 8; ++kk) {
        v16h a = load_afrag(arow, kk * 32, hl);
        v16h b0 = *(const v16h*)(bp + (size_t)kk * 512);
        acc = __builtin_amdgcn_wmma_f32_16x16x32_f16(false, a, false, b0, (short)0, acc, false, false);
    }
    const int n0 = w * 16;
    const float bv = bias[h * DH + n0 + mn];
#pragma unroll
    for (int ri = 0; ri < 8; ++ri) {
        int m = ri + hl * 8;
        out[((r0 + m) * NH + h) * (size_t)DH + n0 + mn] = acc[ri] + bv;
    }
}

// ---------------------------------- LN(og) over (h,d) + sigmoid * cell -> out
__global__ __launch_bounds__(256) void k_ln_og_out(
    const float* __restrict__ og, const float* __restrict__ g,
    const float* __restrict__ bb, const float* __restrict__ cell,
    float* __restrict__ out) {
    __shared__ float rs[256], rq[256];
    int r = blockIdx.x, t = threadIdx.x;
    size_t base = (size_t)r * HD;
    float v[4], s = 0.f, q = 0.f;
#pragma unroll
    for (int j = 0; j < 4; ++j) {
        v[j] = og[base + t + 256 * j];
        s += v[j]; q += v[j] * v[j];
    }
    rs[t] = s; rq[t] = q; __syncthreads();
    for (int st = 128; st; st >>= 1) {
        if (t < st) { rs[t] += rs[t + st]; rq[t] += rq[t + st]; }
        __syncthreads();
    }
    float mean = rs[0] * (1.0f / HD);
    float var  = rq[0] * (1.0f / HD) - mean * mean;
    float rinv = rsqrtf(var + EPS);
#pragma unroll
    for (int j = 0; j < 4; ++j) {
        int flat = t + 256 * j;
        float ln = (v[j] - mean) * rinv * g[flat] + bb[flat];
        out[base + flat] = sigmoidf_(ln) * cell[base + flat];
    }
}

// ---------------------------------------------------------------------------
extern "C" void kernel_launch(void* const* d_in, const int* in_sizes, int n_in,
                              void* d_out, int out_size, void* d_ws, size_t ws_size,
                              hipStream_t stream) {
    const float* x       = (const float*)d_in[0];   // (B,S,H,D)
    const float* W_hid   = (const float*)d_in[1];   // (H,256,384)
    const float* b_hid   = (const float*)d_in[2];   // (H,384)
    const float* W_og    = (const float*)d_in[3];   // (H,256,128)
    const float* b_og    = (const float*)d_in[4];   // (H,128)
    const float* g_cs    = (const float*)d_in[5];
    const float* b_cs    = (const float*)d_in[6];
    const float* g_hid   = (const float*)d_in[7];
    const float* b_hd    = (const float*)d_in[8];
    const float* g_og    = (const float*)d_in[9];
    const float* b_ogl   = (const float*)d_in[10];
    const float* init_cx = (const float*)d_in[11];  // (1,H,D)
    float* out = (float*)d_out;

    // ---- workspace carving (regions reused across pipeline stages) ----
    char* p = (char*)d_ws;
    auto carve = [&](size_t bytes) {
        void* q = (void*)p;
        p += (bytes + 255) & ~(size_t)255;
        return q;
    };
    float*    csum_raw = (float*)carve((size_t)NR * HD * 4);       // later: cell
    _Float16* cat      = (_Float16*)carve((size_t)NR * NH * K2D * 2);
    float*    hid_raw  = (float*)carve((size_t)NR * NH * N3D * 4); // later: og_raw
    float*    fgate    = (float*)carve((size_t)NR * HD * 4);
    float*    igh      = (float*)carve((size_t)NR * HD * 4);
    float*    chunkA   = (float*)carve((size_t)BATCH * NC * HD * 4);
    float*    chunkB   = (float*)carve((size_t)BATCH * NC * HD * 4);
    float*    chunkC   = (float*)carve((size_t)BATCH * NC * HD * 4);
    _Float16* Wh16     = (_Float16*)carve((size_t)NH * K2D * N3D * 2); // swizzled
    _Float16* Wo16     = (_Float16*)carve((size_t)NH * K2D * DH * 2);  // swizzled
    float* cell   = csum_raw;   // reuse: csum dead after k_ln_csum_pack
    float* og_raw = hid_raw;    // reuse: hid dead after k_ln_hid_gates

    const int nWh = NH * K2D * N3D;   // 786432
    const int nWo = NH * K2D * DH;    // 262144

    // 0) weights -> f16, pre-swizzled into WMMA B-fragment order
    k_swz_w<<<(nWh + 255) / 256, 256, 0, stream>>>(W_hid, Wh16, 24, N3D);
    k_swz_w<<<(nWo + 255) / 256, 256, 0, stream>>>(W_og, Wo16, 8, DH);

    // 1) chunked shifted-cumsum over S
    k_csum_chunk<<<512, 256, 0, stream>>>(x, chunkA);
    k_csum_scan<<<16, 256, 0, stream>>>(chunkA, chunkC);
    k_csum_write<<<512, 256, 0, stream>>>(x, chunkC, csum_raw);

    // 2) LN(csum) over (h,d) + pack cat = [f16(x) | f16(LN(csum))]
    k_ln_csum_pack<<<NR, 256, 0, stream>>>(csum_raw, x, g_cs, b_cs, cat);

    // 3) hid = cat @ W_hid + b_hid   (WMMA f16->f32)
    k_hid_gemm<<<dim3(NR / 16, NH), 256, 0, stream>>>(cat, Wh16, b_hid, hid_raw);

    // 4) LN over (h,3,d) + gates
    k_ln_hid_gates<<<NR, 256, 0, stream>>>(hid_raw, g_hid, b_hd, fgate, igh);

    // 5) chunked linear recurrence c = f*c + i; fused f16 repack into cat
    k_cell_chunk<<<512, 256, 0, stream>>>(fgate, igh, chunkA, chunkB);
    k_cell_scan<<<16, 256, 0, stream>>>(chunkA, chunkB, init_cx, chunkC);
    k_cell_write<<<512, 256, 0, stream>>>(fgate, igh, chunkC, cell, cat);

    // 6) og = [x|cell] @ W_og + b_og  (WMMA f16->f32)
    k_og_gemm<<<dim3(NR / 16, NH), 256, 0, stream>>>(cat, Wo16, b_og, og_raw);

    // 7) LN over (h,d) + sigmoid(og)*cell -> out
    k_ln_og_out<<<NR, 256, 0, stream>>>(og_raw, g_og, b_ogl, cell, out);

    (void)in_sizes; (void)n_in; (void)out_size; (void)ws_size;
}
